// NearestEmbed_6390911336467
// MI455X (gfx1250) — matrix operand order repaired and checked by reading, not compiled, not run on hardware
//
#include <hip/hip_runtime.h>

typedef __attribute__((ext_vector_type(16))) __bf16 v16bf;
typedef __attribute__((ext_vector_type(8)))  __bf16 v8bf;
typedef __attribute__((ext_vector_type(8)))  float  v8f;

#define LDSP 72          // padded row pitch (bf16 elems): 144B rows, 16B aligned, bank-spread
#define NCODE 512
#define DDIM 64
#define HWSZ 4096        // H*W
#define BSTRIDE 262144   // D*H*W floats per batch

static __device__ __forceinline__ __bf16 f2bf(float f) {
    unsigned u = __builtin_bit_cast(unsigned, f);
    unsigned r = u + 0x7FFFu + ((u >> 16) & 1u);     // RNE to bf16
    unsigned short s = (unsigned short)(r >> 16);
    return __builtin_bit_cast(__bf16, s);
}
static __device__ __forceinline__ float bf2f(__bf16 h) {
    unsigned short s = __builtin_bit_cast(unsigned short, h);
    unsigned u = ((unsigned)s) << 16;
    return __builtin_bit_cast(float, u);
}

__global__ __launch_bounds__(256)
void NearestEmbed_vq_wmma(const float* __restrict__ x,
                          const float* __restrict__ w,
                          float* __restrict__ out,
                          int* __restrict__ argOut) {
    extern __shared__ char smem[];
    __bf16* sWh = (__bf16*)smem;                 // [512][72] hi bf16
    __bf16* sWl = sWh + NCODE * LDSP;            // [512][72] lo bf16
    float*  sHN = (float*)(sWl + NCODE * LDSP);  // [512] 0.5*||w||^2

    const int tid = threadIdx.x;

    // ---- Stage codebook: transpose (d,k)->[code][d], split fp32 -> bf16 hi/lo, norms ----
    #pragma unroll
    for (int cc = 0; cc < 2; ++cc) {
        const int c = tid + cc * 256;
        float sum = 0.f;
        #pragma unroll
        for (int d = 0; d < DDIM; ++d) {
            float f = w[d * NCODE + c];          // coalesced across threads
            __bf16 h = f2bf(f);
            __bf16 l = f2bf(f - bf2f(h));
            sWh[c * LDSP + d] = h;
            sWl[c * LDSP + d] = l;
            sum += f * f;
        }
        sHN[c] = 0.5f * sum;
    }
    __syncthreads();

    const int lane = tid & 31;
    const int wave = tid >> 5;
    const int tileBase = (blockIdx.x * 8 + wave) * 16;   // 1024 blocks * 8 waves * 16 locs = 131072
    const int loc = tileBase + (lane & 15);              // this lane's location
    const int b  = loc >> 12;
    const int hw = loc & (HWSZ - 1);
    const float* xp = x + (size_t)b * BSTRIDE + hw;

    // ---- Load B fragments (x, K=locations-on-N layout) once; reused for all 32 code tiles ----
    // B k order used here: VGPR v: lanes0-15 -> N=lane, K=2v,2v+1 ; lanes16-31 -> N=lane-16, K=16+2v,17+2v
    v16bf bhf[2], blf[2];
    #pragma unroll
    for (int kh = 0; kh < 2; ++kh) {
        const int kb = ((lane >> 4) << 4) + kh * 32;
        #pragma unroll
        for (int v = 0; v < 8; ++v) {
            float f0 = xp[(kb + 2 * v    ) * HWSZ];
            float f1 = xp[(kb + 2 * v + 1) * HWSZ];
            __bf16 h0 = f2bf(f0), h1 = f2bf(f1);
            bhf[kh][2 * v]     = h0;
            bhf[kh][2 * v + 1] = h1;
            blf[kh][2 * v]     = f2bf(f0 - bf2f(h0));
            blf[kh][2 * v + 1] = f2bf(f1 - bf2f(h1));
        }
    }

    float bestv = -3.0e38f;
    int   besti = 0;
    const int arow = lane & 15;            // A (code) row this lane serves
    const int aoff = (lane >> 4) << 4;     // K sub-offset 0/16 per A layout
    const int cadd = (lane >> 4) << 3;     // C layout: upper lanes hold codes M=j+8

    for (int ct = 0; ct < 32; ++ct) {
        const __bf16* rh = sWh + (ct * 16 + arow) * LDSP;
        const __bf16* rl = sWl + (ct * 16 + arow) * LDSP;
        v8f acc0 = {};
        v8f acc1 = {};
        #pragma unroll
        for (int kh = 0; kh < 2; ++kh) {
            const int o = kh * 32 + aoff;
            // A k order used here: VGPR0-7 = K o..o+15 contiguous (2 per dword), matching B
            v16bf ah = __builtin_shufflevector(*(const v8bf*)(rh + o),
                                               *(const v8bf*)(rh + o + 8),
                                               0,1,2,3,4,5,6,7,8,9,10,11,12,13,14,15);
            v16bf al = __builtin_shufflevector(*(const v8bf*)(rl + o),
                                               *(const v8bf*)(rl + o + 8),
                                               0,1,2,3,4,5,6,7,8,9,10,11,12,13,14,15);
            // score = xh*wh + xh*wl + xl*wh  (lo*lo dropped)
            acc0 = __builtin_amdgcn_wmma_f32_16x16x32_bf16(false, ah, false, bhf[kh],
                                                           (short)0, acc0, false, false);
            acc1 = __builtin_amdgcn_wmma_f32_16x16x32_bf16(false, ah, false, blf[kh],
                                                           (short)0, acc1, false, false);
            acc0 = __builtin_amdgcn_wmma_f32_16x16x32_bf16(false, al, false, bhf[kh],
                                                           (short)0, acc0, false, false);
        }
        // C layout: VGPR j, lane n<16 -> (code ct*16+j, loc n); lane>=16 -> (code ct*16+j+8, loc n-16)
        #pragma unroll
        for (int j = 0; j < 8; ++j) {
            const int code = ct * 16 + j + cadd;
            float s = acc0[j] + acc1[j] - sHN[code];   // argmax of dot - 0.5||w||^2 == argmin dist
            if (s > bestv) { bestv = s; besti = code; }
        }
    }

    // lanes n and n+16 hold complementary code halves for the same location: combine
    {
        float ov = __shfl_xor(bestv, 16, 32);
        int   oi = __shfl_xor(besti, 16, 32);
        if (ov > bestv || (ov == bestv && oi < besti)) { bestv = ov; besti = oi; }
    }

    if (lane < 16) argOut[loc] = besti;

    // ---- Gather quantized output (exact fp32 codebook), coalesced over hw ----
    float* op = out + (size_t)b * BSTRIDE + hw;
    const int dBase = (lane >> 4) << 5;    // lanes 0-15 write d=0..31, lanes 16-31 write d=32..63
    #pragma unroll
    for (int dd = 0; dd < 32; ++dd) {
        const int d = dBase + dd;
        op[d * HWSZ] = w[d * NCODE + besti];
    }
}

extern "C" void kernel_launch(void* const* d_in, const int* in_sizes, int n_in,
                              void* d_out, int out_size, void* d_ws, size_t ws_size,
                              hipStream_t stream) {
    (void)in_sizes; (void)n_in; (void)out_size; (void)d_ws; (void)ws_size;
    const float* x = (const float*)d_in[0];
    const float* w = (const float*)d_in[1];
    float* out = (float*)d_out;
    int* argOut = (int*)(out + (size_t)32 * 64 * 64 * 64);   // argmin follows result in d_out

    const size_t shmem = (size_t)NCODE * LDSP * 2u * 2u + (size_t)NCODE * 4u;  // ~146 KB
    hipFuncSetAttribute(reinterpret_cast<const void*>(NearestEmbed_vq_wmma),
                        hipFuncAttributeMaxDynamicSharedMemorySize, (int)shmem);
    NearestEmbed_vq_wmma<<<1024, 256, shmem, stream>>>(x, w, out, argOut);
}
